// SparseUNet_84155589198509
// MI455X (gfx1250) — matrix operand order repaired and checked by reading, not compiled, not run on hardware
//
#include <hip/hip_runtime.h>
#include <cstddef>

// ---------------------------------------------------------------------------
// Types for WMMA
// ---------------------------------------------------------------------------
typedef __attribute__((ext_vector_type(16))) _Float16 v16h;
typedef __attribute__((ext_vector_type(8)))  _Float16 h8;
typedef __attribute__((ext_vector_type(8)))  float    v8f;

// ---------------------------------------------------------------------------
// Weight pack: W[k][ci][co] (f32) -> Wp in per-lane WMMA-B layout (f16):
//   Wp[(((k*nch + ch)*ctiles + ct)*32 + lane)*16 + e] = W[k][ch*32 + 16*hi + e][ct*16 + ln]
// (hi = lane>>4, ln = lane&15; zero-filled past Ci). One contiguous 32B per lane.
// ---------------------------------------------------------------------------
__global__ __launch_bounds__(256) void pack_w_kernel(
    const float* __restrict__ W, _Float16* __restrict__ Wp,
    int K, int Ci, int Co, int nch)
{
  const int ctiles = Co >> 4;
  const size_t total = (size_t)K * nch * ctiles * 512;
  for (size_t i = (size_t)blockIdx.x * blockDim.x + threadIdx.x; i < total;
       i += (size_t)gridDim.x * blockDim.x) {
    const int e    = (int)(i & 15);
    const int lane = (int)((i >> 4) & 31);
    size_t rest    = i >> 9;
    const int ct = (int)(rest % ctiles); rest /= ctiles;
    const int ch = (int)(rest % nch);
    const int k  = (int)(rest / nch);
    const int hi = lane >> 4, ln = lane & 15;
    const int ci = ch * 32 + 16 * hi + e;
    const int co = ct * 16 + ln;
    const float v = (ci < Ci) ? W[((size_t)k * Ci + ci) * Co + co] : 0.f;
    Wp[i] = (_Float16)v;
  }
}

// ---------------------------------------------------------------------------
// Sparse conv: out[rout[k,p], co] += sum_ci f[rin[k,p], ci] * W[k, ci, co]
// One wave32 per (k, 64-row group, CT co-tiles): 4 row-tiles x CT co-tiles
// share A gathers and B loads -> 4*CT independent v_wmma_f32_16x16x32_f16 per
// 32-channel chunk. f16 gather (2x b128/row), f32 accumulate, f32 atomic
// scatter. Pads: rin pad=0 (safe gather), rout pad=n_out (dummy row).
// ---------------------------------------------------------------------------
template <int CT>
__global__ __launch_bounds__(256) void conv_wmma_kernel(
    const _Float16* __restrict__ f, const _Float16* __restrict__ Wp,
    const int* __restrict__ rin, const int* __restrict__ rout,
    float* __restrict__ out,
    int K, int P, int Ci, int Co, int n_out)
{
  const int wid  = blockIdx.x * (blockDim.x >> 5) + (threadIdx.x >> 5);
  const int lane = threadIdx.x & 31;
  const int ctiles = Co >> 4;
  const int ctgn   = ctiles / CT;              // groups of CT co-tiles
  const int nch    = (Ci + 31) >> 5;
  const int pgn    = (P + 63) >> 6;            // groups of 4 row-tiles
  const int total  = K * pgn * ctgn;
  if (wid >= total) return;                    // wave-uniform exit

  const int k   = wid / (pgn * ctgn);
  const int r2  = wid - k * (pgn * ctgn);
  const int pg  = r2 / ctgn;
  const int ctg = r2 - pg * ctgn;
  const int hi  = lane >> 4;
  const int ln  = lane & 15;

  // gather base rows for 4 row-tiles (this lane owns matrix row M = ln)
  const _Float16* frow[4];
  #pragma unroll
  for (int m = 0; m < 4; ++m) {
    const int p = pg * 64 + m * 16 + ln;
    const int r = (p < P) ? rin[k * P + p] : 0;
    frow[m] = f + (size_t)r * Ci;
  }

  v8f acc[CT][4];
  #pragma unroll
  for (int c = 0; c < CT; ++c)
    #pragma unroll
    for (int m = 0; m < 4; ++m) acc[c][m] = (v8f){};

  const size_t wstep = (size_t)ctiles * 512;   // halves per (k,ch) plane
  const _Float16* wptr =
      Wp + (((size_t)(k * nch) * ctiles + (size_t)ctg * CT) * 32 + lane) * 16;
  const bool aligned = ((Ci & 7) == 0);

  for (int ch = 0; ch < nch; ++ch, wptr += wstep) {
    v16h b[CT];
    #pragma unroll
    for (int c = 0; c < CT; ++c)
      b[c] = *(const v16h*)(wptr + (size_t)c * 512); // 32B contiguous each

    const int s0 = ch * 32 + 16 * hi;          // A elems 0..7  -> ci = s0..s0+7
    const int s1 = s0 + 8;                     // A elems 8..15 -> ci = s1..s1+7
    #pragma unroll
    for (int m = 0; m < 4; ++m) {
      h8 lo = {}, hi8 = {};
      if (aligned) {
        if (s0 < Ci) lo  = *(const h8*)(frow[m] + s0);
        if (s1 < Ci) hi8 = *(const h8*)(frow[m] + s1);
      } else {                                 // ragged (Ci=3 stem)
        #pragma unroll
        for (int e = 0; e < 8; ++e) {
          lo[e]  = (s0 + e < Ci) ? frow[m][s0 + e] : (_Float16)0.f;
          hi8[e] = (s1 + e < Ci) ? frow[m][s1 + e] : (_Float16)0.f;
        }
      }
      const v16h a = __builtin_shufflevector(
          lo, hi8, 0, 1, 2, 3, 4, 5, 6, 7, 8, 9, 10, 11, 12, 13, 14, 15);
      #pragma unroll
      for (int c = 0; c < CT; ++c)
        acc[c][m] = __builtin_amdgcn_wmma_f32_16x16x32_f16(
            false, a, false, b[c], (short)0, acc[c][m], false, false);
    }
  }

  // scatter-add: D VGPR r holds (M = r + 8*hi, N = ln); rows shared across CT
  float* obase = out + (size_t)(ctg * CT) * 16 + ln;
  #pragma unroll
  for (int m = 0; m < 4; ++m) {
    #pragma unroll
    for (int r = 0; r < 8; ++r) {
      const int M = r + 8 * hi;
      const int p = pg * 64 + m * 16 + M;
      const int orow = (p < P) ? rout[k * P + p] : n_out;
      float* orowp = obase + (size_t)orow * Co;
      #pragma unroll
      for (int c = 0; c < CT; ++c)
        unsafeAtomicAdd(orowp + (size_t)c * 16, acc[c][m][r]);
    }
  }
}

// ---------------------------------------------------------------------------
// f32 -> f16 conversion (stem features)
// ---------------------------------------------------------------------------
__global__ __launch_bounds__(256) void cvt_kernel(
    const float* __restrict__ src, _Float16* __restrict__ dst, size_t count)
{
  for (size_t i = (size_t)blockIdx.x * blockDim.x + threadIdx.x; i < count;
       i += (size_t)gridDim.x * blockDim.x)
    dst[i] = (_Float16)src[i];
}

// ---------------------------------------------------------------------------
// BatchNorm stats: per-channel sum & sumsq (LDS partials, then global atomics)
// ---------------------------------------------------------------------------
__global__ __launch_bounds__(256) void bn_stats_kernel(
    const float* __restrict__ x, int n, int C, float* __restrict__ stats)
{
  __shared__ float ssum[256];
  __shared__ float ssq[256];
  for (int c = threadIdx.x; c < C; c += blockDim.x) { ssum[c] = 0.f; ssq[c] = 0.f; }
  __syncthreads();
  const size_t total = (size_t)n * C;
  for (size_t i = (size_t)blockIdx.x * blockDim.x + threadIdx.x; i < total;
       i += (size_t)gridDim.x * blockDim.x) {
    const int c = (int)(i % C);
    const float v = x[i];
    atomicAdd(&ssum[c], v);
    atomicAdd(&ssq[c], v * v);
  }
  __syncthreads();
  for (int c = threadIdx.x; c < C; c += blockDim.x) {
    atomicAdd(&stats[c], ssum[c]);
    atomicAdd(&stats[C + c], ssq[c]);
  }
}

// ---------------------------------------------------------------------------
// BN apply (+optional f16 residual, +ReLU); writes f32 and/or f16 outputs
// ---------------------------------------------------------------------------
__global__ __launch_bounds__(256) void bn_apply_kernel(
    const float* __restrict__ x, const float* __restrict__ stats,
    const float* __restrict__ g, const float* __restrict__ b,
    const _Float16* __restrict__ res,
    float* __restrict__ out32, _Float16* __restrict__ out16,
    int n, int C, int relu)
{
  const float inv_n = 1.f / (float)n;
  const size_t total = (size_t)n * C;
  for (size_t i = (size_t)blockIdx.x * blockDim.x + threadIdx.x; i < total;
       i += (size_t)gridDim.x * blockDim.x) {
    const int c = (int)(i % C);
    const float mean = stats[c] * inv_n;
    const float var  = stats[C + c] * inv_n - mean * mean;   // biased var
    float v = (x[i] - mean) * __frsqrt_rn(var + 1e-5f) * g[c] + b[c];
    if (res) v += (float)res[i];
    if (relu) v = fmaxf(v, 0.f);
    if (out32) out32[i] = v;
    if (out16) out16[i] = (_Float16)v;
  }
}

// ---------------------------------------------------------------------------
// Channel concat (f16): out[i] = [a | b]
// ---------------------------------------------------------------------------
__global__ __launch_bounds__(256) void concat_kernel(
    const _Float16* __restrict__ a, const _Float16* __restrict__ b,
    _Float16* __restrict__ out, int n, int ca, int cb)
{
  const int C = ca + cb;
  const size_t total = (size_t)n * C;
  for (size_t i = (size_t)blockIdx.x * blockDim.x + threadIdx.x; i < total;
       i += (size_t)gridDim.x * blockDim.x) {
    const int c = (int)(i % C);
    const size_t r = i / C;
    out[i] = (c < ca) ? a[r * (size_t)ca + c] : b[r * (size_t)cb + (c - ca)];
  }
}

// ---------------------------------------------------------------------------
// Host orchestration
// ---------------------------------------------------------------------------
static inline int cdiv_i(int a, int b) { return (a + b - 1) / b; }
static inline int grid_cap(size_t total) {
  long long bl = (long long)((total + 255) / 256);
  if (bl > 4096) bl = 4096;
  if (bl < 1) bl = 1;
  return (int)bl;
}

extern "C" void kernel_launch(void* const* d_in, const int* in_sizes, int n_in,
                              void* d_out, int out_size, void* d_ws, size_t ws_size,
                              hipStream_t stream)
{
  // flattened input map (setup_inputs dict order):
  //   0 features, 1 coords, [2..100] params (99), [101..132] rules (32), meta ignored
  const int PB = 2, RB = 101;
  if (n_in < RB + 32) return;

  const float* feats = (const float*)d_in[0];
  const int N0 = in_sizes[0] / 3;

  auto pw   = [&](int i) -> const float* { return (const float*)d_in[PB + i]; };
  auto ridx = [&](int i) -> const int*   { return (const int*)d_in[RB + i]; };

  // rules: RB+0/1 = subm0; per level t: RB+2+6t = down in/out, invin/invout, subm in/out
  int Nlev[5], Pdown[5], Pinv[5];
  for (int t = 0; t < 5; ++t) {
    Pdown[t] = in_sizes[RB + 2 + 6 * t + 0] / 8;
    Pinv[t]  = in_sizes[RB + 2 + 6 * t + 2] / 8;
    Nlev[t]  = in_sizes[RB + 2 + 6 * t + 4] / 27;  // subm center offset full => P == N
  }
  static const int CHh[5] = {16, 32, 64, 128, 256};

  // deterministic bump allocator over d_ws
  char* wp = (char*)d_ws;
  auto alloc32 = [&](size_t elems) -> float* {
    float* p = (float*)wp;
    wp += (elems * sizeof(float) + 255) & ~(size_t)255;
    return p;
  };
  auto alloc16 = [&](size_t elems) -> _Float16* {
    _Float16* p = (_Float16*)wp;
    wp += (elems * sizeof(_Float16) + 255) & ~(size_t)255;
    return p;
  };

  float* stats = alloc32(512);

  auto do_conv = [&](const _Float16* fh, int Ci, const float* W32,
                     const int* rin, const int* rout, int K, int P,
                     int n, float* out, int Co) {
    const int nch = (Ci + 31) / 32, ctiles = Co / 16;
    const size_t pk = (size_t)K * nch * ctiles * 512;
    _Float16* Wp = alloc16(pk);
    pack_w_kernel<<<grid_cap(pk), 256, 0, stream>>>(W32, Wp, K, Ci, Co, nch);
    hipMemsetAsync(out, 0, (size_t)(n + 1) * Co * sizeof(float), stream);
    const int CT = (ctiles % 2 == 0) ? 2 : 1;
    const int waves  = K * cdiv_i(P, 64) * (ctiles / CT);
    const int blocks = cdiv_i(waves, 8);
    if (CT == 2)
      conv_wmma_kernel<2><<<blocks, 256, 0, stream>>>(fh, Wp, rin, rout, out,
                                                      K, P, Ci, Co, n);
    else
      conv_wmma_kernel<1><<<blocks, 256, 0, stream>>>(fh, Wp, rin, rout, out,
                                                      K, P, Ci, Co, n);
  };

  auto do_bn = [&](const float* x, int n, int C, const float* g, const float* b,
                   const _Float16* res, float* out32, _Float16* out16) {
    hipMemsetAsync(stats, 0, 2 * (size_t)C * sizeof(float), stream);
    const int blocks = grid_cap((size_t)n * C);
    bn_stats_kernel<<<blocks, 256, 0, stream>>>(x, n, C, stats);
    bn_apply_kernel<<<blocks, 256, 0, stream>>>(x, stats, g, b, res, out32,
                                                out16, n, C, 1);
  };

  // ---- stem: features f32 -> f16, subm conv (3 -> 16) at level '0' ----
  _Float16* featsh = alloc16((size_t)N0 * 3);
  cvt_kernel<<<grid_cap((size_t)N0 * 3), 256, 0, stream>>>(feats, featsh,
                                                           (size_t)N0 * 3);
  float* S0 = alloc32((size_t)(N0 + 1) * 16);
  _Float16* x0h = alloc16((size_t)(N0 + 1) * 16);
  do_conv(featsh, 3, pw(0), ridx(0), ridx(1), 27, N0, N0, S0, 16);
  do_bn(S0, N0, 16, pw(1), pw(2), nullptr, nullptr, x0h);

  // per-level buffers: f32 conv scratch S, f16 activations Ah/Bh/Ch, Dh (2C)
  float* S[5];
  _Float16 *Ah[5], *Bh[5], *Ch[5], *Dh[5];
  for (int t = 0; t < 5; ++t) {
    const size_t re = (size_t)(Nlev[t] + 1) * CHh[t];
    S[t]  = alloc32(re);
    Ah[t] = alloc16(re);
    Bh[t] = alloc16(re);
    Ch[t] = alloc16(re);
    Dh[t] = alloc16(re * 2);
  }

  // ---- encoder ----
  const _Float16* xh = x0h;
  int ci = 16;
  for (int t = 0; t < 5; ++t) {
    const int oc = CHh[t], n = Nlev[t];
    const int* sin  = ridx(2 + 6 * t + 4);
    const int* sout = ridx(2 + 6 * t + 5);
    do_conv(xh, ci, pw(3 + 9 * t + 0), ridx(2 + 6 * t + 0), ridx(2 + 6 * t + 1),
            8, Pdown[t], n, S[t], oc);
    do_bn(S[t], n, oc, pw(3 + 9 * t + 1), pw(3 + 9 * t + 2), nullptr, nullptr, Ah[t]);
    do_conv(Ah[t], oc, pw(3 + 9 * t + 3), sin, sout, 27, n, n, S[t], oc);
    do_bn(S[t], n, oc, pw(3 + 9 * t + 4), pw(3 + 9 * t + 5), nullptr, nullptr, Bh[t]);
    do_conv(Bh[t], oc, pw(3 + 9 * t + 6), sin, sout, 27, n, n, S[t], oc);
    do_bn(S[t], n, oc, pw(3 + 9 * t + 7), pw(3 + 9 * t + 8), Ah[t], nullptr, Ch[t]);
    xh = Ch[t];                     // Ch[t] persists as encoder skip
    ci = oc;
  }

  // ---- decoder ----
  for (int d = 0; d < 4; ++d) {
    const int s = 4 - d, t = s - 1;
    const int oc = CHh[t], n = Nlev[t];
    const int* sin  = ridx(2 + 6 * t + 4);
    const int* sout = ridx(2 + 6 * t + 5);
    const int pb = 48 + 12 * d;
    // inverse conv of down{s} (K=8): level s -> level t
    do_conv(xh, ci, pw(pb + 0), ridx(2 + 6 * s + 2), ridx(2 + 6 * s + 3),
            8, Pinv[s], n, S[t], oc);
    do_bn(S[t], n, oc, pw(pb + 1), pw(pb + 2), nullptr, nullptr, Ah[t]);
    // concat [up | enc skip]
    concat_kernel<<<grid_cap((size_t)n * 2 * oc), 256, 0, stream>>>(
        Ah[t], Ch[t], Dh[t], n, oc, oc);
    // upcat conv (2*oc -> oc)
    do_conv(Dh[t], 2 * oc, pw(pb + 3), sin, sout, 27, n, n, S[t], oc);
    do_bn(S[t], n, oc, pw(pb + 4), pw(pb + 5), nullptr, nullptr, Bh[t]);
    // residual block (h1 -> Ah, h2 -> Dh reuse, res = Bh)
    do_conv(Bh[t], oc, pw(pb + 6), sin, sout, 27, n, n, S[t], oc);
    do_bn(S[t], n, oc, pw(pb + 7), pw(pb + 8), nullptr, nullptr, Ah[t]);
    do_conv(Ah[t], oc, pw(pb + 9), sin, sout, 27, n, n, S[t], oc);
    do_bn(S[t], n, oc, pw(pb + 10), pw(pb + 11), Bh[t], nullptr, Dh[t]);
    xh = Dh[t];
    ci = oc;
  }

  // ---- output head at level '0e' ----
  {
    const int n = Nlev[0];
    do_conv(xh, 16, pw(96), ridx(2 + 4), ridx(2 + 5), 27, n, n, S[0], 16);
    do_bn(S[0], n, 16, pw(97), pw(98), nullptr, (float*)d_out, nullptr);
  }
}
